// SubjectSpecificLayer_57715770523928
// MI455X (gfx1250) — matrix operand order repaired and checked by reading, not compiled, not run on hardware
//
#include <hip/hip_runtime.h>

// ---------------------------------------------------------------------------
// Per-sample expert GEMM: out[b] = X[b] @ W[subject[b]] + bias[subject[b]]
// X: [B,281,272] f32, subject: [B] i64, W: [4,272,272] f32, b: [4,272] f32
//
// MI455X strategy:
//  - Memory-bound (~68 FLOP/B, HBM floor ~54us at 23.3 TB/s). W (1.2MB) is
//    L2-resident noise; X read + out write dominate.
//  - Split-precision bf16 WMMA: x=xh+xl, w=wh+wl; acc += xh*wh + xh*wl + xl*wh
//    (v_wmma_f32_16x16x32_bf16, ~fp32 accuracy at bf16 matrix rate).
//  - PREP kernel converts W once to transposed, K-padded (288) bf16 hi/lo in
//    d_ws. Main kernel streams W chunks LDS-ward with the Tensor Data Mover
//    (tensor_load_to_lds), double-buffered, using TDM padding to produce the
//    40-elem LDS pitch the WMMA fragment loads expect. Sync: s_wait_tensorcnt.
//  - 544-thread block (17 wave32): wave j owns N-strip j (272 = 17*16);
//    block computes 64 rows x 272 cols of one sample.
// ---------------------------------------------------------------------------

typedef __attribute__((ext_vector_type(16))) __bf16        v16bf;
typedef __attribute__((ext_vector_type(8)))  float         v8f;
typedef __attribute__((ext_vector_type(4)))  unsigned int  v4u;
typedef __attribute__((ext_vector_type(8)))  int           v8i;
typedef __attribute__((ext_vector_type(4)))  int           v4i;

constexpr int T_DIM    = 281;
constexpr int D        = 272;                 // 17*16
constexpr int KPAD     = 288;                 // K padded to 9*32
constexpr int BM       = 64;                  // M rows per block
constexpr int KC       = 32;                  // K chunk
constexpr int NKC      = KPAD / KC;           // 9
constexpr int XP       = 40;                  // X LDS pitch (elems, 80B)
constexpr int WP       = 40;                  // W^T LDS pitch (elems, 80B)
constexpr int NWAVES   = 17;
constexpr int NTHREADS = NWAVES * 32;         // 544
constexpr int NSUBJ    = 4;
constexpr int WT_ELEMS = NSUBJ * D * KPAD;    // elems per (hi|lo) part

__device__ __forceinline__ unsigned short f2bf(float f) {
  unsigned int u = __builtin_bit_cast(unsigned int, f);
  u += 0x7fffu + ((u >> 16) & 1u);            // RNE
  return (unsigned short)(u >> 16);
}
__device__ __forceinline__ float bf2f(unsigned short h) {
  unsigned int u = ((unsigned int)h) << 16;
  return __builtin_bit_cast(float, u);
}

union BFV { v16bf v; unsigned short u[16]; };

// B fragment (32x16 bf16): 16 consecutive K at fixed column n = 32 contiguous
// bytes in the [n][k] LDS image -> 2x ds_load_b128.
__device__ __forceinline__ v16bf load_b_frag(const unsigned short* p) {
  BFV r;
#pragma unroll
  for (int i = 0; i < 16; ++i) r.u[i] = p[i];
  return r.v;
}
// A fragment (16x32 bf16): element i = K=half*16+i (same order as B).
__device__ __forceinline__ v16bf load_a_frag(const unsigned short* row, int half) {
  BFV r;
  const unsigned short* p0 = row + half * 16;
#pragma unroll
  for (int i = 0; i < 16; ++i) { r.u[i] = p0[i]; }
  return r.v;
}

// Issue a TDM load of a 32(k) x 272(n) tile of the transposed expert matrix
// (row length KPAD bf16) into LDS with pitch WP=40 elems (pad 4 DWORDs after
// every 16 DWORDs = one 64B tile row).
__device__ __forceinline__ void tdm_load_w(const unsigned short* gsrc,
                                           unsigned short* lds_dst) {
  unsigned long long ga = (unsigned long long)(size_t)gsrc;
  unsigned int la = (unsigned int)(size_t)(void*)lds_dst;  // LDS addr = low 32b
  v4u g0;
  g0[0] = 1u;                                        // count=1 (valid), no gather
  g0[1] = la;                                        // lds_addr
  g0[2] = (unsigned int)ga;                          // global_addr[31:0]
  g0[3] = (unsigned int)((ga >> 32) & 0x01FFFFFFu)   // global_addr[56:32]
          | (2u << 30);                              // type=2 ("image")
  v8i g1;
  g1[0] = (int)((1u << 16)      // data_size = 2 bytes
              | (1u << 20)      // pad_enable
              | (3u << 22)      // pad_interval: 16 DWORDs (64B) between pads
              | (3u << 25));    // pad_amount: 4 DWORDs (16B)
  g1[1] = (int)(((unsigned)KPAD & 0xFFFFu) << 16);   // tensor_dim0 lo16 @bit48
  g1[2] = (int)(((unsigned)KPAD >> 16) | ((unsigned)D << 16)); // dim0 hi | dim1 lo
  g1[3] = (int)(((unsigned)D >> 16) | ((unsigned)KC << 16));   // dim1 hi | tile_dim0
  g1[4] = (int)((unsigned)D);                        // tile_dim1=272, tile_dim2=0
  g1[5] = (int)((unsigned)KPAD);                     // tensor_dim0_stride lo32
  g1[6] = 0;                                         // stride0 hi | stride1 lo
  g1[7] = 0;
  v4i z4 = {0, 0, 0, 0};
  v8i z8 = {0, 0, 0, 0, 0, 0, 0, 0};
  // amdgpu-toolchain (clang-23) 6-arg form:
  // (uint32x4 g0, int32x8 g1, int32x4, int32x4, int32x8, i32 cpol)
  __builtin_amdgcn_tensor_load_to_lds(g0, g1, z4, z4, z8, 0);
}

// ---- prep: W[e][k][n] f32 -> wt_hi/wt_lo[e][n][k] bf16, K zero-padded to 288
__global__ void prep_w(const float* __restrict__ W,
                       unsigned short* __restrict__ wt_hi,
                       unsigned short* __restrict__ wt_lo) {
  int idx = blockIdx.x * 256 + threadIdx.x;
  if (idx >= WT_ELEMS) return;
  int k  = idx % KPAD;
  int en = idx / KPAD;                 // e*D + n
  int n  = en % D;
  int e  = en / D;
  float v = (k < D) ? W[((size_t)e * D + k) * D + n] : 0.0f;
  unsigned short h = f2bf(v);
  unsigned short l = f2bf(v - bf2f(h));
  wt_hi[idx] = h;
  wt_lo[idx] = l;
}

__global__ __launch_bounds__(NTHREADS)
void subject_moe_gemm(const float* __restrict__ X,
                      const long long* __restrict__ subj,
                      const unsigned short* __restrict__ wt_hi,
                      const unsigned short* __restrict__ wt_lo,
                      const float* __restrict__ bias,
                      float* __restrict__ out) {
  __shared__ unsigned short xh[BM * XP];
  __shared__ unsigned short xl[BM * XP];
  __shared__ unsigned short wbh[2][D * WP];   // double-buffered W^T hi
  __shared__ unsigned short wbl[2][D * WP];   // double-buffered W^T lo

  const int b    = blockIdx.x;
  const int m0   = blockIdx.y * BM;
  const int e    = (int)subj[b];
  const int tid  = threadIdx.x;
  const int lane = tid & 31;
  const int wave = tid >> 5;
  const int half = lane >> 4;
  const int ln   = lane & 15;

  const float* Xb = X + (size_t)b * T_DIM * D;
  const unsigned short* wsh = wt_hi + (size_t)e * D * KPAD;
  const unsigned short* wsl = wt_lo + (size_t)e * D * KPAD;

  v8f acc[4] = {};

  // prefetch W chunk 0 via TDM (wave 0 only; one DMA per part)
  if (wave == 0) {
    tdm_load_w(wsh, &wbh[0][0]);
    tdm_load_w(wsl, &wbl[0][0]);
  }

  for (int kc = 0; kc < NKC; ++kc) {
    const int k0  = kc * KC;
    const int cur = kc & 1;

    // async prefetch of next W chunk into the other buffer
    if (wave == 0 && kc + 1 < NKC) {
      tdm_load_w(wsh + (size_t)(kc + 1) * KC, &wbh[cur ^ 1][0]);
      tdm_load_w(wsl + (size_t)(kc + 1) * KC, &wbl[cur ^ 1][0]);
    }

    // stage X chunk (fp32 -> split bf16), 64x32 elems
#pragma unroll
    for (int i = 0; i < 4; ++i) {
      int idx = tid + i * NTHREADS;
      if (idx < BM * KC) {
        int r   = idx >> 5;
        int k   = idx & 31;
        int row = m0 + r;
        float v = (row < T_DIM && (k0 + k) < D) ? Xb[(size_t)row * D + k0 + k]
                                                : 0.0f;
        unsigned short h = f2bf(v);
        unsigned short l = f2bf(v - bf2f(h));
        xh[r * XP + k] = h;
        xl[r * XP + k] = l;
      }
    }

    // ensure chunk kc's two TDM loads are complete (per-wave in-order):
    // with chunk kc+1 just issued, "<=2" retires exactly the older pair.
    if (wave == 0) {
      if (kc + 1 < NKC) __builtin_amdgcn_s_wait_tensorcnt(2);
      else              __builtin_amdgcn_s_wait_tensorcnt(0);
    }
    __syncthreads();

    const int ncol = wave * 16 + ln;
    const int kb   = half * 16;
    v16bf bfh = load_b_frag(&wbh[cur][ncol * WP + kb]);
    v16bf bfl = load_b_frag(&wbl[cur][ncol * WP + kb]);

#pragma unroll
    for (int mt = 0; mt < 4; ++mt) {
      const unsigned short* rh = &xh[(mt * 16 + ln) * XP];
      const unsigned short* rl = &xl[(mt * 16 + ln) * XP];
      v16bf afh = load_a_frag(rh, half);
      v16bf afl = load_a_frag(rl, half);
      acc[mt] = __builtin_amdgcn_wmma_f32_16x16x32_bf16(
          false, afh, false, bfh, (short)0, acc[mt], false, false);
      acc[mt] = __builtin_amdgcn_wmma_f32_16x16x32_bf16(
          false, afh, false, bfl, (short)0, acc[mt], false, false);
      acc[mt] = __builtin_amdgcn_wmma_f32_16x16x32_bf16(
          false, afl, false, bfh, (short)0, acc[mt], false, false);
    }
    __syncthreads();   // protect xh/xl (and retired W buffer) before restage
  }

  // epilogue: bias + store (C layout: VGPR r -> M = r + 8*half, N = ln)
  const int ncol = wave * 16 + ln;
  const float bv = bias[(size_t)e * D + ncol];
  float* Ob = out + (size_t)b * T_DIM * D;
#pragma unroll
  for (int mt = 0; mt < 4; ++mt) {
#pragma unroll
    for (int r = 0; r < 8; ++r) {
      int M = m0 + mt * 16 + r + half * 8;
      if (M < T_DIM) Ob[(size_t)M * D + ncol] = acc[mt][r] + bv;
    }
  }
}

extern "C" void kernel_launch(void* const* d_in, const int* in_sizes, int n_in,
                              void* d_out, int out_size, void* d_ws, size_t ws_size,
                              hipStream_t stream) {
  const float*     X    = (const float*)d_in[0];
  const long long* subj = (const long long*)d_in[1];   // jnp int64
  const float*     W    = (const float*)d_in[2];
  const float*     bias = (const float*)d_in[3];
  float*           out  = (float*)d_out;

  unsigned short* wt_hi = (unsigned short*)d_ws;       // [4][272][288]
  unsigned short* wt_lo = wt_hi + WT_ELEMS;            // ~1.25MB total

  const int B = in_sizes[0] / (T_DIM * D);             // 2048

  prep_w<<<(WT_ELEMS + 255) / 256, 256, 0, stream>>>(W, wt_hi, wt_lo);

  dim3 grid(B, (T_DIM + BM - 1) / BM);                 // 2048 x 5
  subject_moe_gemm<<<grid, NTHREADS, 0, stream>>>(X, subj, wt_hi, wt_lo,
                                                  bias, out);
}
